// CustomRNN_manual_17454747091430
// MI455X (gfx1250) — matrix operand order repaired and checked by reading, not compiled, not run on hardware
//
#include <hip/hip_runtime.h>
#include <hip/hip_bf16.h>

// CDNA5 / gfx1250 RNN kernel.
// Decomposition:
//   1) convert weights + embedding table to bf16 (workspace)
//   2) xproj[t,b,:] = emb[tok[b,t]] @ Wx^T + b_i2h   (parallel bf16 WMMA GEMM)
//   3) persistent cooperative recurrence kernel: per step
//        pre = xproj[t] + h @ Wh^T   (bf16 WMMA, f32 accum, acc lives in regs)
//        LN stats via shfl-xor reduce + global_atomic_add_f32
//        grid barrier; LN+tanh+residual+clip; write h (f32) + h (bf16)
//        grid barrier
//   4) out = h @ Who^T + b_h2o (bf16 WMMA) ; copy hidden f32

typedef __attribute__((ext_vector_type(16))) __bf16 v16bf;
typedef __attribute__((ext_vector_type(8)))  __bf16 v8bf;
typedef __attribute__((ext_vector_type(8)))  float  v8f;

static constexpr int kVocab = 50257;
static constexpr int kEmb   = 512;
static constexpr int kHid   = 1024;
static constexpr int kOut   = 1024;
static constexpr int kB     = 128;
static constexpr int kT     = 512;
static constexpr int kRecurBlocks = 64;   // 8 row-tiles x 8 col-chunks
#define LN_EPS 1e-5f

__device__ __forceinline__ unsigned short f2bf_raw(float f) {
  unsigned u = __builtin_bit_cast(unsigned, f);
  unsigned r = u + 0x7FFFu + ((u >> 16) & 1u);   // round-to-nearest-even
  return (unsigned short)(r >> 16);
}

// A fragment (16x32 bf16, M x K): lane m = lane&15, half = lane>>4.
// frag[0..15] <- K = k0 + half*16 + (0..15)
__device__ __forceinline__ v16bf load_a_frag(const __bf16* __restrict__ row, int k0, int half) {
  return *(const v16bf*)(row + k0 + half * 16);
}

// B fragment (32x16 bf16, K x N): lane n = lane&15, half = lane>>4.
// frag[0..15] <- K = k0 + half*16 + (0..15)  (contiguous run of W row n)
__device__ __forceinline__ v16bf load_b_frag(const __bf16* __restrict__ row, int k0, int half) {
  return *(const v16bf*)(row + k0 + half * 16);
}

__device__ __forceinline__ v8f wmma_bf16(v16bf a, v16bf b, v8f c) {
  return __builtin_amdgcn_wmma_f32_16x16x32_bf16(false, a, false, b, (short)0, c, false, false);
}

// Device-wide software barrier (monotonic counter, reset by init kernel per launch).
__device__ __forceinline__ void grid_sync(unsigned* bar, unsigned target) {
  __threadfence();                 // release my stores to device scope
  __syncthreads();
  if (threadIdx.x == 0) {
    atomicAdd(bar, 1u);
    while (__hip_atomic_load(bar, __ATOMIC_ACQUIRE, __HIP_MEMORY_SCOPE_AGENT) < target) {
      __builtin_amdgcn_s_sleep(2);
    }
  }
  __syncthreads();
  __threadfence();                 // acquire: invalidate near caches
}

// ---------------------------------------------------------------- conversion
__global__ void __launch_bounds__(256) k_cvt_bf16(const float* __restrict__ src,
                                                  unsigned short* __restrict__ dst,
                                                  int cols, int srcStride, int srcOff, long n) {
  long i = (long)blockIdx.x * blockDim.x + threadIdx.x;
  if (i >= n) return;
  long r = i / cols;
  int  c = (int)(i - r * cols);
  dst[i] = f2bf_raw(src[r * srcStride + srcOff + c]);
}

// ---------------------------------------------------------------- init state
__global__ void __launch_bounds__(256) k_init(const float* __restrict__ h0,
                                              float* __restrict__ h,
                                              unsigned short* __restrict__ h_bf,
                                              float* __restrict__ stats,
                                              unsigned* __restrict__ bar) {
  int i = blockIdx.x * blockDim.x + threadIdx.x;
  if (i < kB * kHid) {
    float v = h0[i];
    h[i] = v;
    h_bf[i] = f2bf_raw(v);
  }
  if (i < 512) stats[i] = 0.0f;   // two ping-pong buffers of (sum[128], sumsq[128])
  if (i == 0) *bar = 0u;
}

// --------------------------------------------------- xproj = emb @ Wx^T + b
__global__ void __launch_bounds__(256) k_xproj(const int* __restrict__ tokens,
                                               const unsigned short* __restrict__ emb_bf,
                                               const unsigned short* __restrict__ wx_bf,
                                               const float* __restrict__ bias,
                                               float* __restrict__ xproj) {
  int wid  = blockIdx.x * (blockDim.x >> 5) + (threadIdx.x >> 5);
  int lane = threadIdx.x & 31;
  int t  = wid >> 9;
  int rem = wid & 511;
  int tm = rem >> 6;      // batch tile (8)
  int tn = rem & 63;      // output tile (64)
  int r16 = lane & 15, half = lane >> 4;

  int b   = tm * 16 + r16;
  int tok = tokens[b * kT + t];
  const __bf16* arow = (const __bf16*)emb_bf + (long)tok * kEmb;
  const __bf16* brow = (const __bf16*)wx_bf + (long)(tn * 16 + r16) * kEmb;

  v8f acc = {};
#pragma unroll 4
  for (int k0 = 0; k0 < kEmb; k0 += 32) {
    v16bf a  = load_a_frag(arow, k0, half);
    v16bf bb = load_b_frag(brow, k0, half);
    acc = wmma_bf16(a, bb, acc);
  }

  int col = tn * 16 + r16;
  float bv = bias[col];
  long base = ((long)t * kB + tm * 16) * kHid + col;
#pragma unroll
  for (int r = 0; r < 8; ++r) {
    int M = r + 8 * half;
    xproj[base + (long)M * kHid] = acc[r] + bv;
  }
}

// ------------------------------------------------- persistent recurrence
__global__ void __launch_bounds__(256, 1) k_recur(const float* __restrict__ xproj,
                                                  const unsigned short* __restrict__ wh_bf,
                                                  const float* __restrict__ gamma,
                                                  const float* __restrict__ beta,
                                                  float* __restrict__ h,
                                                  unsigned short* __restrict__ h_bf,
                                                  float* __restrict__ stats,
                                                  unsigned* __restrict__ bar) {
  int tmb   = blockIdx.x & 7;         // row tile (16 batch rows)
  int chunk = blockIdx.x >> 3;        // col chunk (128 cols)
  int wave  = threadIdx.x >> 5;
  int lane  = threadIdx.x & 31;
  int tn    = chunk * 8 + wave;       // this wave's 16-col tile
  int r16 = lane & 15, half = lane >> 4;

  int arow_idx = tmb * 16 + r16;
  int col      = tn * 16 + r16;
  const __bf16* brow = (const __bf16*)wh_bf + (long)col * kHid;

  for (int t = 0; t < kT; ++t) {
    // zero the *other* stats buffer for step t+1 (safe: its readers finished
    // before barrier 2t, its writers start after barrier 2t+2)
    if (chunk == 0 && threadIdx.x < 16) {
      int m = tmb * 16 + threadIdx.x;
      stats[(((t + 1) & 1) << 8) + m]       = 0.0f;
      stats[(((t + 1) & 1) << 8) + 128 + m] = 0.0f;
    }

    // --- phase 1: pre-tile = h @ Wh^T + xproj[t], accum stays in registers
    const __bf16* arow = (const __bf16*)h_bf + (long)arow_idx * kHid;
    v8f acc = {};
#pragma unroll 4
    for (int k0 = 0; k0 < kHid; k0 += 32) {
      v16bf a  = load_a_frag(arow, k0, half);
      v16bf bb = load_b_frag(brow, k0, half);
      acc = wmma_bf16(a, bb, acc);
    }

    long xb = ((long)t * kB + tmb * 16) * kHid + col;
    if (t + 1 < kT) __builtin_prefetch(xproj + xb + (long)kB * kHid, 0, 1);

    float pre[8];
#pragma unroll
    for (int r = 0; r < 8; ++r)
      pre[r] = acc[r] + xproj[xb + (long)(r + 8 * half) * kHid];

    // --- row stats: butterfly reduce across the 16 lanes of this half-group
    float* ssum = stats + ((t & 1) << 8);
#pragma unroll
    for (int r = 0; r < 8; ++r) {
      float sv = pre[r];
      float qv = pre[r] * pre[r];
#pragma unroll
      for (int m = 1; m < 16; m <<= 1) {
        sv += __shfl_xor(sv, m, 32);
        qv += __shfl_xor(qv, m, 32);
      }
      if (r16 == 0) {
        int m = tmb * 16 + 8 * half + r;
        atomicAdd(&ssum[m], sv);
        atomicAdd(&ssum[128 + m], qv);
      }
    }

    grid_sync(bar, (unsigned)kRecurBlocks * (2u * t + 1u));

    // --- phase 2: LayerNorm + tanh + residual + clip, update h (f32 + bf16)
    float gm = gamma[col], bt = beta[col];
#pragma unroll
    for (int r = 0; r < 8; ++r) {
      int m = tmb * 16 + 8 * half + r;
      float mu   = ssum[m] * (1.0f / kHid);
      float var  = ssum[128 + m] * (1.0f / kHid) - mu * mu;
      float rstd = rsqrtf(var + LN_EPS);
      float ln   = (pre[r] - mu) * rstd * gm + bt;
      float nh   = tanhf(ln);
      long idx   = (long)m * kHid + col;
      float hv   = 0.5f * h[idx] + nh;
      hv = fminf(5.0f, fmaxf(-5.0f, hv));
      h[idx]    = hv;
      h_bf[idx] = f2bf_raw(hv);
    }

    grid_sync(bar, (unsigned)kRecurBlocks * (2u * t + 2u));
  }
}

// ------------------------------------------------- out = h @ Who^T + b_h2o
__global__ void __launch_bounds__(256) k_out(const unsigned short* __restrict__ h_bf,
                                             const unsigned short* __restrict__ who_bf,
                                             const float* __restrict__ bias,
                                             float* __restrict__ out) {
  int wid  = blockIdx.x * (blockDim.x >> 5) + (threadIdx.x >> 5);  // 512 waves
  int lane = threadIdx.x & 31;
  int tm = wid >> 6, tn = wid & 63;
  int r16 = lane & 15, half = lane >> 4;

  const __bf16* arow = (const __bf16*)h_bf + (long)(tm * 16 + r16) * kHid;
  const __bf16* brow = (const __bf16*)who_bf + (long)(tn * 16 + r16) * kHid;

  v8f acc = {};
#pragma unroll 4
  for (int k0 = 0; k0 < kHid; k0 += 32) {
    v16bf a  = load_a_frag(arow, k0, half);
    v16bf bb = load_b_frag(brow, k0, half);
    acc = wmma_bf16(a, bb, acc);
  }

  int col = tn * 16 + r16;
  float bv = bias[col];
#pragma unroll
  for (int r = 0; r < 8; ++r)
    out[(long)(tm * 16 + r + 8 * half) * kOut + col] = acc[r] + bv;
}

__global__ void __launch_bounds__(256) k_copy_hidden(const float* __restrict__ h,
                                                     float* __restrict__ dst) {
  int i = blockIdx.x * blockDim.x + threadIdx.x;
  if (i < kB * kHid) dst[i] = h[i];
}

// --------------------------------------------------------------------------
extern "C" void kernel_launch(void* const* d_in, const int* in_sizes, int n_in,
                              void* d_out, int out_size, void* d_ws, size_t ws_size,
                              hipStream_t stream) {
  (void)in_sizes; (void)n_in; (void)out_size; (void)ws_size;
  const int*   tokens = (const int*)d_in[0];     // (B, T)
  const float* h0     = (const float*)d_in[1];   // (B, HID)
  const float* emb    = (const float*)d_in[2];   // (VOCAB, EMB)
  const float* W_i2h  = (const float*)d_in[3];   // (HID, EMB+HID)
  const float* b_i2h  = (const float*)d_in[4];   // (HID)
  const float* gamma  = (const float*)d_in[5];   // (HID)
  const float* beta   = (const float*)d_in[6];   // (HID)
  const float* W_h2o  = (const float*)d_in[7];   // (OUT, HID)
  const float* b_h2o  = (const float*)d_in[8];   // (OUT)
  float* out = (float*)d_out;                    // [output(128x1024) | hidden(128x1024)]

  // workspace carve (256B aligned slices)
  char* p = (char*)d_ws;
  auto take = [&](size_t bytes) -> char* {
    char* r = p;
    p += (bytes + 255) & ~(size_t)255;
    return r;
  };
  unsigned short* wx_bf  = (unsigned short*)take((size_t)kHid * kEmb * 2);
  unsigned short* wh_bf  = (unsigned short*)take((size_t)kHid * kHid * 2);
  unsigned short* who_bf = (unsigned short*)take((size_t)kOut * kHid * 2);
  unsigned short* emb_bf = (unsigned short*)take((size_t)kVocab * kEmb * 2);
  float*          h      = (float*)take((size_t)kB * kHid * 4);
  unsigned short* h_bf   = (unsigned short*)take((size_t)kB * kHid * 2);
  float*          stats  = (float*)take(512 * 4);
  unsigned*       bar    = (unsigned*)take(256);
  float*          xproj  = (float*)take((size_t)kT * kB * kHid * 4);

  auto blocks = [](long n) { return (unsigned)((n + 255) / 256); };

  // 1) bf16 weight / embedding conversion
  { long n = (long)kHid * kEmb;
    k_cvt_bf16<<<blocks(n), 256, 0, stream>>>(W_i2h, wx_bf, kEmb, kEmb + kHid, 0, n); }
  { long n = (long)kHid * kHid;
    k_cvt_bf16<<<blocks(n), 256, 0, stream>>>(W_i2h, wh_bf, kHid, kEmb + kHid, kEmb, n); }
  { long n = (long)kOut * kHid;
    k_cvt_bf16<<<blocks(n), 256, 0, stream>>>(W_h2o, who_bf, kHid, kHid, 0, n); }
  { long n = (long)kVocab * kEmb;
    k_cvt_bf16<<<blocks(n), 256, 0, stream>>>(emb, emb_bf, kEmb, kEmb, 0, n); }

  // 2) state init (h, h_bf, LN stats, barrier counter) — every launch, so
  //    graph replay is deterministic
  k_init<<<blocks((long)kB * kHid), 256, 0, stream>>>(h0, h, h_bf, stats, bar);

  // 3) parallel x-projection: T*8*64 = 262144 tiles, 8 waves/block
  k_xproj<<<(kT * 512) / 8, 256, 0, stream>>>(tokens, emb_bf, wx_bf, b_i2h, xproj);

  // 4) persistent cooperative recurrence (64 blocks, device barriers)
  k_recur<<<kRecurBlocks, 256, 0, stream>>>(xproj, wh_bf, gamma, beta, h, h_bf, stats, bar);

  // 5) output projection + hidden copy
  k_out<<<kRecurBlocks, 256, 0, stream>>>(h_bf, who_bf, b_h2o, out);
  k_copy_hidden<<<blocks((long)kB * kHid), 256, 0, stream>>>(h, out + (size_t)kB * kHid);
}
